// JDFlow_56238301773943
// MI455X (gfx1250) — compile-verified, matrix operand order checked by reading, not run on hardware
//
#include <hip/hip_runtime.h>
#include <hip/hip_bf16.h>

typedef __attribute__((ext_vector_type(16))) _Float16 v16h;
typedef __attribute__((ext_vector_type(8)))  _Float16 v8h;
typedef __attribute__((ext_vector_type(8)))  float    v8f;

#define M_DIM 1024
#define T_DIM 1024
#define H_DIM 2048
#define D_DIM 1024
#define HID   128
#define G_DIM 4096
#define DT_C  (1.0f/1024.0f)

__device__ __forceinline__ float sigmoidf_(float x) { return 1.0f / (1.0f + __expf(-x)); }

// Low 32 bits of a generic pointer into __shared__ = LDS byte offset
__device__ __forceinline__ unsigned lds_addr32_(const void* p) {
  return (unsigned)(size_t)p;
}

__device__ __forceinline__ void async_wait0_() {
#if __has_builtin(__builtin_amdgcn_s_wait_asynccnt)
  __builtin_amdgcn_s_wait_asynccnt(0);
#else
  asm volatile("s_wait_asynccnt 0x0" ::: "memory");
#endif
}

// Async DMA one 64-byte row slice global -> LDS (no VGPR staging, ASYNCcnt)
__device__ __forceinline__ void async_copy64_(unsigned ldsa, const void* gsrc) {
  asm volatile(
      "global_load_async_to_lds_b128 %0, %1, off\n\t"
      "global_load_async_to_lds_b128 %0, %1, off offset:16\n\t"
      "global_load_async_to_lds_b128 %0, %1, off offset:32\n\t"
      "global_load_async_to_lds_b128 %0, %1, off offset:48"
      :: "v"(ldsa), "v"(gsrc) : "memory");
}

// ---------------------------------------------------------------------------
// WMMA f16 GEMM:  C[M x N] = epi(A[M x K] * B[K x N] + bias),  B given as
// Bt[N x K] row-major (pre-transposed).  Tile 128x128x32, 256 threads (8 waves).
// EPI: 0=linear 1=sigmoid 2=tanh 3=exp(-x)
// ---------------------------------------------------------------------------
__device__ __forceinline__ v16h ldfrag32(const _Float16* base, int rt, int lane) {
  // 16-bit A-matrix 16x32 layout: lanes 0-15 -> row, K 0..7 & 16..23;
  // lanes 16-31 -> row, K 8..15 & 24..31 (BK = 32 halves per row in LDS)
  int row = rt + (lane & 15);
  int kb  = (lane >> 4) << 3;          // 0 or 8
  const v8h* p = (const v8h*)(base + row * 32 + kb);
  v8h lo = p[0];                        // K = kb .. kb+7
  v8h hi = p[2];                        // K = kb+16 .. kb+23
  return __builtin_shufflevector(lo, hi, 0,1,2,3,4,5,6,7,8,9,10,11,12,13,14,15);
}

// Issue async tile load: 128 rows of A and 128 rows of Bt, 64B per thread
__device__ __forceinline__ void load_tile32_async(const _Float16* __restrict__ A,
                                                  const _Float16* __restrict__ Bt,
                                                  _Float16* as, _Float16* bs,
                                                  int rowA0, int rowB0, int kb, int Kdim, int tid) {
  if (tid < 128) {
    async_copy64_(lds_addr32_(as + tid * 32),
                  (const void*)(A + (size_t)(rowA0 + tid) * Kdim + kb));
  } else {
    int r = tid - 128;
    async_copy64_(lds_addr32_(bs + r * 32),
                  (const void*)(Bt + (size_t)(rowB0 + r) * Kdim + kb));
  }
}

template<int EPI>
__global__ __launch_bounds__(256) void gemm_f16_wmma(const _Float16* __restrict__ A,
                                                     const _Float16* __restrict__ Bt,
                                                     const float* __restrict__ bias,
                                                     float* __restrict__ Cout,
                                                     int Mdim, int Ndim, int Kdim) {
  __shared__ __align__(16) _Float16 As[2][128 * 32];
  __shared__ __align__(16) _Float16 Bs[2][128 * 32];
  const int tid  = threadIdx.x;
  const int lane = tid & 31;
  const int wave = tid >> 5;
  const int m0 = (wave >> 1) * 32;      // wave M base within tile (4 rows of waves)
  const int n0 = (wave & 1) * 64;       // wave N base within tile (2 cols of waves)
  const int rowA0 = blockIdx.x * 128;
  const int rowB0 = blockIdx.y * 128;

  v8f acc[2][4] = {};
  const int nk = Kdim >> 5;

  load_tile32_async(A, Bt, As[0], Bs[0], rowA0, rowB0, 0, Kdim, tid);
  async_wait0_();
  __syncthreads();

  for (int kt = 0; kt < nk; ++kt) {
    const int buf = kt & 1;
    if (kt + 1 < nk)
      load_tile32_async(A, Bt, As[buf ^ 1], Bs[buf ^ 1], rowA0, rowB0, (kt + 1) << 5, Kdim, tid);
    if (kt + 2 < nk) {
      const int kb2 = (kt + 2) << 5;
      if (tid < 128) __builtin_prefetch(A  + (size_t)(rowA0 + tid)       * Kdim + kb2, 0, 1);
      else           __builtin_prefetch(Bt + (size_t)(rowB0 + tid - 128) * Kdim + kb2, 0, 1);
    }

    v16h af[2], bf[4];
    af[0] = ldfrag32(As[buf], m0,      lane);
    af[1] = ldfrag32(As[buf], m0 + 16, lane);
#pragma unroll
    for (int j = 0; j < 4; ++j) bf[j] = ldfrag32(Bs[buf], n0 + j * 16, lane);
#pragma unroll
    for (int i = 0; i < 2; ++i)
#pragma unroll
      for (int j = 0; j < 4; ++j)
        acc[i][j] = __builtin_amdgcn_wmma_f32_16x16x32_f16(
            false, af[i], false, bf[j], (short)0, acc[i][j], false, false);

    async_wait0_();      // next tile fully landed in LDS
    __syncthreads();
  }

  const int row0 = rowA0 + m0;
  const int col0 = rowB0 + n0;
#pragma unroll
  for (int j = 0; j < 4; ++j) {
    const int n = col0 + j * 16 + (lane & 15);
    const float bv = bias ? bias[n] : 0.0f;
#pragma unroll
    for (int i = 0; i < 2; ++i) {
      const int mbase = row0 + i * 16 + ((lane >> 4) << 3);
#pragma unroll
      for (int r = 0; r < 8; ++r) {
        float v = acc[i][j][r] + bv;
        if (EPI == 1) v = sigmoidf_(v);
        else if (EPI == 2) v = tanhf(v);
        else if (EPI == 3) v = __expf(-v);
        Cout[(size_t)(mbase + r) * Ndim + n] = v;
      }
    }
  }
}

// ---------------------------------------------------------------------------
// Packing / elementwise kernels
// ---------------------------------------------------------------------------
template<int OP>  // 0 = copy, 1 = sigmoid
__global__ void pack16_mat_kernel(_Float16* dst, const float* src, int rows, int cols, int stride) {
  int idx = blockIdx.x * 256 + threadIdx.x;
  if (idx >= rows * cols) return;
  int r = idx / cols, c = idx - r * cols;
  float v = src[(size_t)r * stride + c];
  if (OP == 1) v = sigmoidf_(v);
  dst[idx] = (_Float16)v;
}

// dst[r*C + c] = src[c*stride + r]   (dst R x C, src C x R with row stride)
__global__ void pack16_t_kernel(_Float16* dst, const float* src, int R, int C, int stride) {
  int idx = blockIdx.x * 256 + threadIdx.x;
  if (idx >= R * C) return;
  int r = idx / C, c = idx - r * C;
  dst[idx] = (_Float16)src[(size_t)c * stride + r];
}

__global__ void addvec_kernel(float* c, const float* a, const float* b, int n) {
  int i = blockIdx.x * 256 + threadIdx.x;
  if (i < n) c[i] = a[i] + b[i];
}

__global__ void zero_kernel(float* p, int n) {
  int i = blockIdx.x * 256 + threadIdx.x;
  if (i < n) p[i] = 0.0f;
}

__global__ void combine_nl_kernel(float* xbuf, const float* nl, const float* x, const float* zo, int n) {
  int i = blockIdx.x * 256 + threadIdx.x;
  if (i < n) xbuf[i] = nl[i] * x[i] + zo[i];
}

__global__ void add_half_kernel(float* xbuf, const float* phat, int c1) {
  int idx = blockIdx.x * 256 + threadIdx.x;
  if (idx >= M_DIM * D_DIM) return;
  int m = idx >> 10, d = idx & 1023;
  xbuf[(size_t)m * H_DIM + c1 + d] += phat[idx];
}

__global__ void flow_out_kernel(float* xbuf, const float* mu, const float* fv,
                                const float* sig, int c2) {
  int idx = blockIdx.x * 256 + threadIdx.x;
  if (idx >= M_DIM * D_DIM) return;
  int m = idx >> 10, d = idx & 1023;
  size_t o = (size_t)m * H_DIM + c2 + d;
  xbuf[o] = mu[idx] * fv[idx] + xbuf[o] * __expf(sig[idx]);
}

__global__ void rowsum_add_kernel(float* dst, const float* src, int cols) {
  __shared__ float red[256];
  int row = blockIdx.x;
  float s = 0.0f;
  for (int c = threadIdx.x; c < cols; c += 256) s += src[(size_t)row * cols + c];
  red[threadIdx.x] = s; __syncthreads();
  for (int off = 128; off > 0; off >>= 1) {
    if (threadIdx.x < off) red[threadIdx.x] += red[threadIdx.x + off];
    __syncthreads();
  }
  if (threadIdx.x == 0) dst[row] += red[0];
}

__global__ void logpz_kernel(float* dst, const float* xbuf) {
  __shared__ float red[256];
  int row = blockIdx.x;
  float s = 0.0f;
  for (int c = threadIdx.x; c < H_DIM; c += 256) {
    float v = xbuf[(size_t)row * H_DIM + c];
    s += v * v;
  }
  red[threadIdx.x] = s; __syncthreads();
  for (int off = 128; off > 0; off >>= 1) {
    if (threadIdx.x < off) red[threadIdx.x] += red[threadIdx.x + off];
    __syncthreads();
  }
  if (threadIdx.x == 0)
    dst[row] = -0.5f * red[0] - 0.5f * (float)H_DIM * 1.8378770664093453f;
}

// ---------------------------------------------------------------------------
// v0 = mlp_sig(phi, concat([wt, x[:,0]]))  — single workgroup matvec
// ---------------------------------------------------------------------------
__global__ __launch_bounds__(1024) void phi_v0_kernel(const float* wt, const float* x,
                                                      const float* w1, const float* b1,
                                                      const float* w2, const float* b2,
                                                      float* vt0) {
  __shared__ float in[2048];
  __shared__ float h[HID];
  int tid = threadIdx.x;
  in[tid] = wt[tid];
  in[1024 + tid] = x[(size_t)tid * H_DIM];
  __syncthreads();
  if (tid < HID) {
    float s = b1[tid];
    for (int i = 0; i < 2048; ++i) s += in[i] * w1[i * HID + tid];
    h[tid] = sigmoidf_(s);
  }
  __syncthreads();
  float s = b2[tid];
  for (int j = 0; j < HID; ++j) s += h[j] * w2[j * M_DIM + tid];
  vt0[tid] = s;
}

// ---------------------------------------------------------------------------
// SDE jump-diffusion scan — persistent single workgroup of 1024 threads
// ---------------------------------------------------------------------------
__global__ __launch_bounds__(1024) void sde_scan_kernel(
    const float* dr_w1, const float* dr_b1, const float* dr_w2, const float* dr_b2,
    const float* di_w1, const float* di_b1, const float* di_w2, const float* di_b2,
    const float* ju_w1, const float* ju_b1, const float* ju_w2, const float* ju_b2,
    const float* dW, const float* dN, float* vt) {
  __shared__ float v[M_DIM];
  __shared__ float hsh[3 * HID];
  __shared__ float diffs;
  int tid = threadIdx.x;
  v[tid] = vt[tid];
  __syncthreads();
  for (int t = 0; t < T_DIM - 1; ++t) {
    if (tid < 3 * HID) {
      int grp = tid >> 7, j = tid & 127;
      const float* w1 = (grp == 0) ? dr_w1 : (grp == 1) ? di_w1 : ju_w1;
      const float* b1 = (grp == 0) ? dr_b1 : (grp == 1) ? di_b1 : ju_b1;
      float s = b1[j];
      for (int i = 0; i < M_DIM; ++i) s += v[i] * w1[i * HID + j];
      hsh[tid] = sigmoidf_(s);
    }
    __syncthreads();
    if (tid == 0) {
      float s = di_b2[0];
      for (int j = 0; j < HID; ++j) s += hsh[HID + j] * di_w2[j];
      diffs = s;
    }
    __syncthreads();
    float dr = dr_b2[tid], ju = ju_b2[tid];
    for (int j = 0; j < HID; ++j) {
      dr += hsh[j] * dr_w2[j * M_DIM + tid];
      ju += hsh[2 * HID + j] * ju_w2[j * M_DIM + tid];
    }
    float nv = v[tid] + dr * DT_C + diffs * dW[(size_t)t * M_DIM + tid]
                      + ju * dN[(size_t)t * M_DIM + tid];
    __syncthreads();
    v[tid] = nv;
    vt[(size_t)(t + 1) * M_DIM + tid] = nv;
    __syncthreads();
  }
}

// ---------------------------------------------------------------------------
// LSTM scan — pre = xs@wih + (bih+bhh) done by WMMA GEMM; this does the
// recurrent h@whh part.  whhT is [4096 x 1024] f16 (row = gate).
// ---------------------------------------------------------------------------
__global__ __launch_bounds__(1024) void lstm_scan_kernel(const float* __restrict__ pre,
                                                         const _Float16* __restrict__ whhT,
                                                         float* __restrict__ hs) {
  __shared__ float h[D_DIM];
  __shared__ float c[D_DIM];
  __shared__ float g[G_DIM];
  int tid = threadIdx.x;
  h[tid] = 0.0f; c[tid] = 0.0f;
  __syncthreads();
  for (int t = 0; t < M_DIM; ++t) {
#pragma unroll
    for (int q = 0; q < 4; ++q) {
      int gidx = q * D_DIM + tid;
      const _Float16* wr = whhT + (size_t)gidx * D_DIM;
      float s = pre[(size_t)t * G_DIM + gidx];
      for (int k = 0; k < D_DIM; ++k) s += h[k] * (float)wr[k];
      g[gidx] = s;
    }
    __syncthreads();
    float iv = sigmoidf_(g[tid]);
    float fv = sigmoidf_(g[D_DIM + tid]);
    float gv = tanhf(g[2 * D_DIM + tid]);
    float ov = sigmoidf_(g[3 * D_DIM + tid]);
    float cn = fv * c[tid] + iv * gv;
    float hn = ov * tanhf(cn);
    __syncthreads();
    c[tid] = cn; h[tid] = hn;
    hs[(size_t)t * D_DIM + tid] = hn;
    __syncthreads();
  }
}

// ---------------------------------------------------------------------------
// Host orchestration
// ---------------------------------------------------------------------------
static inline void* bump_(char*& p, size_t bytes) {
  void* r = (void*)p;
  p += (bytes + 255) & ~(size_t)255;
  return r;
}

extern "C" void kernel_launch(void* const* d_in, const int* in_sizes, int n_in,
                              void* d_out, int out_size, void* d_ws, size_t ws_size,
                              hipStream_t stream) {
  (void)in_sizes; (void)n_in; (void)out_size; (void)ws_size;

  // --- input index map (dict/tree flatten: alphabetical keys, 'b' < 'w') ---
  // 0:x | diff 1-4 | drift 5-8 | flows 9+32*i | jump 201-204 | phi 205-208 |
  // psi 209-214 | zeta 215-220 | 221:wt 222:dW 223:dN
  auto F32 = [&](int i) { return (const float*)d_in[i]; };
  const int DIFF_B0 = 1, DIFF_W0 = 2, DIFF_B1 = 3, DIFF_W1 = 4;
  const int DRIFT_B0 = 5, DRIFT_W0 = 6, DRIFT_B1 = 7, DRIFT_W1 = 8;
  const int JUMP_B0 = 201, JUMP_W0 = 202, JUMP_B1 = 203, JUMP_W1 = 204;
  const int PHI_B0 = 205, PHI_W0 = 206, PHI_B1 = 207, PHI_W1 = 208;
  const int PSI = 209, ZETA = 215, IDX_WT = 221, IDX_DW = 222, IDX_DN = 223;

  // --- output regions ---
  float* out_x      = (float*)d_out;
  float* out_logpz  = out_x + (size_t)M_DIM * H_DIM;
  float* out_logjac = out_logpz + M_DIM;
  float* vt         = out_logjac + M_DIM;            // T x M

  // --- workspace ---
  char* wp = (char*)d_ws;
  _Float16* vtT16 = (_Float16*)bump_(wp, (size_t)M_DIM * T_DIM * 2);
  float*    hf32  = (float*)   bump_(wp, (size_t)M_DIM * HID * 4);
  _Float16* h16   = (_Float16*)bump_(wp, (size_t)M_DIM * HID * 2);
  _Float16* wA    = (_Float16*)bump_(wp, (size_t)HID * H_DIM * 2);   // up to 128x2048
  _Float16* wB    = (_Float16*)bump_(wp, (size_t)HID * HID * 2);
  _Float16* wC    = (_Float16*)bump_(wp, (size_t)H_DIM * HID * 2);   // up to 2048x128
  float*    nl    = (float*)   bump_(wp, (size_t)M_DIM * H_DIM * 4);
  _Float16* a16b  = (_Float16*)bump_(wp, (size_t)M_DIM * H_DIM * 2); // nl packed
  float*    zo    = (float*)   bump_(wp, (size_t)M_DIM * H_DIM * 4);
  float*    xbuf  = (float*)   bump_(wp, (size_t)M_DIM * H_DIM * 4);
  _Float16* z1_16 = (_Float16*)bump_(wp, (size_t)M_DIM * D_DIM * 2);
  _Float16* z1T16 = (_Float16*)bump_(wp, (size_t)D_DIM * M_DIM * 2);
  _Float16* sh16  = (_Float16*)bump_(wp, (size_t)M_DIM * D_DIM * 2);
  _Float16* nu16  = (_Float16*)bump_(wp, (size_t)M_DIM * D_DIM * 2);
  float*    t0    = (float*)   bump_(wp, (size_t)M_DIM * D_DIM * 4);
  float*    t1    = (float*)   bump_(wp, (size_t)M_DIM * D_DIM * 4);
  float*    t2    = (float*)   bump_(wp, (size_t)M_DIM * D_DIM * 4);
  float*    t3    = (float*)   bump_(wp, (size_t)M_DIM * D_DIM * 4);
  float*    pre   = (float*)   bump_(wp, (size_t)M_DIM * G_DIM * 4);
  float*    hsbuf = (float*)   bump_(wp, (size_t)M_DIM * D_DIM * 4);
  _Float16* wBig1 = (_Float16*)bump_(wp, (size_t)G_DIM * D_DIM * 2);
  _Float16* wBig2 = (_Float16*)bump_(wp, (size_t)G_DIM * D_DIM * 2);
  float*    b4096 = (float*)   bump_(wp, (size_t)G_DIM * 4);

  // --- helpers ---
  auto gemm = [&](int epi, const _Float16* A, const _Float16* Bt, const float* bias,
                  float* C, int Mi, int Ni, int Ki) {
    dim3 g(Mi / 128, Ni / 128), b(256);
    if (epi == 0)      gemm_f16_wmma<0><<<g, b, 0, stream>>>(A, Bt, bias, C, Mi, Ni, Ki);
    else if (epi == 1) gemm_f16_wmma<1><<<g, b, 0, stream>>>(A, Bt, bias, C, Mi, Ni, Ki);
    else if (epi == 2) gemm_f16_wmma<2><<<g, b, 0, stream>>>(A, Bt, bias, C, Mi, Ni, Ki);
    else               gemm_f16_wmma<3><<<g, b, 0, stream>>>(A, Bt, bias, C, Mi, Ni, Ki);
  };
  auto packT = [&](_Float16* dst, const float* src, int R, int C, int stride) {
    int n = R * C;
    pack16_t_kernel<<<(n + 255) / 256, 256, 0, stream>>>(dst, src, R, C, stride);
  };
  auto packWT = [&](_Float16* dst, const float* w, int K, int N) { packT(dst, w, N, K, N); };
  auto packM = [&](int op, _Float16* dst, const float* src, int rows, int cols, int stride) {
    int n = rows * cols;
    dim3 g((n + 255) / 256), b(256);
    if (op == 0) pack16_mat_kernel<0><<<g, b, 0, stream>>>(dst, src, rows, cols, stride);
    else         pack16_mat_kernel<1><<<g, b, 0, stream>>>(dst, src, rows, cols, stride);
  };
  // 3-layer mlp_sig: K0 -> 128 -> 128 -> Nout, final epilogue epiF
  auto mlp3 = [&](const _Float16* A16, int K0, int Nout,
                  const float* b0, const float* w0, const float* b1, const float* w1,
                  const float* b2, const float* w2, int epiF, float* out) {
    packWT(wA, w0, K0, HID);
    gemm(1, A16, wA, b0, hf32, M_DIM, HID, K0);
    packM(0, h16, hf32, M_DIM, HID, HID);
    packWT(wB, w1, HID, HID);
    gemm(1, h16, wB, b1, hf32, M_DIM, HID, HID);
    packM(0, h16, hf32, M_DIM, HID, HID);
    packWT(wC, w2, HID, Nout);
    gemm(epiF, h16, wC, b2, out, M_DIM, Nout, HID);
  };

  // --- 1. init log_jacob ---
  zero_kernel<<<(M_DIM + 255) / 256, 256, 0, stream>>>(out_logjac, M_DIM);

  // --- 2. v0 = mlp_sig(phi, [wt, x[:,0]]) -> vt row 0 ---
  phi_v0_kernel<<<1, 1024, 0, stream>>>(F32(IDX_WT), F32(0),
      F32(PHI_W0), F32(PHI_B0), F32(PHI_W1), F32(PHI_B1), vt);

  // --- 3. SDE scan -> vt rows 1..T-1 ---
  sde_scan_kernel<<<1, 1024, 0, stream>>>(
      F32(DRIFT_W0), F32(DRIFT_B0), F32(DRIFT_W1), F32(DRIFT_B1),
      F32(DIFF_W0),  F32(DIFF_B0),  F32(DIFF_W1),  F32(DIFF_B1),
      F32(JUMP_W0),  F32(JUMP_B0),  F32(JUMP_W1),  F32(JUMP_B1),
      F32(IDX_DW), F32(IDX_DN), vt);

  // --- 4. vt.T packed once (shared by psi and every flow's p-MLP) ---
  packT(vtT16, vt, M_DIM, T_DIM, M_DIM);

  // --- 5. nl = exp(-mlp_sig(psi, vt.T)) ---
  mlp3(vtT16, T_DIM, H_DIM,
       F32(PSI + 0), F32(PSI + 1), F32(PSI + 2), F32(PSI + 3), F32(PSI + 4), F32(PSI + 5),
       3, nl);

  // --- 6. x = nl*x + mlp_sig(zeta, nl) ---
  packM(0, a16b, nl, M_DIM, H_DIM, H_DIM);
  mlp3(a16b, H_DIM, H_DIM,
       F32(ZETA + 0), F32(ZETA + 1), F32(ZETA + 2), F32(ZETA + 3), F32(ZETA + 4), F32(ZETA + 5),
       0, zo);
  {
    int n = M_DIM * H_DIM;
    combine_nl_kernel<<<(n + 255) / 256, 256, 0, stream>>>(xbuf, nl, F32(0), zo, n);
  }

  // --- 7. flows ---
  const int NHALF = M_DIM * D_DIM;
  for (int fi = 0; fi < 6; ++fi) {
    const int base = 9 + fi * 32;
    const int F_B0 = base + 0,  F_W0 = base + 1,  F_B1 = base + 2,  F_W1 = base + 3;
    const int MU   = base + 4;   // b0,w0,b1,w1,b2,w2
    const int SG   = base + 10;
    const int BHH  = base + 16, BIH = base + 17, WHH = base + 18, WIH = base + 19;
    const int NU   = base + 20;
    const int PP   = base + 26;
    const int c1 = (fi & 1) ? D_DIM : 0;   // half that x1/z1 lives in
    const int c2 = D_DIM - c1;             // half that x2/z2 lives in

    // p_hat = tanh(mlp_sig(p, vt.T));  z1 = x1 + p_hat  (in place on xbuf)
    mlp3(vtT16, T_DIM, D_DIM,
         F32(PP + 0), F32(PP + 1), F32(PP + 2), F32(PP + 3), F32(PP + 4), F32(PP + 5),
         2, t0);
    add_half_kernel<<<(NHALF + 255) / 256, 256, 0, stream>>>(xbuf, t0, c1);

    // pack z1 (row-major) and z1.T
    packM(0, z1_16, xbuf + c1, M_DIM, D_DIM, H_DIM);
    packT(z1T16, xbuf + c1, D_DIM, M_DIM, H_DIM);

    // LSTM: pre = z1 @ wih + (bih + bhh)   (WMMA GEMM), then recurrent scan
    addvec_kernel<<<(G_DIM + 255) / 256, 256, 0, stream>>>(b4096, F32(BIH), F32(BHH), G_DIM);
    packT(wBig1, F32(WIH), G_DIM, D_DIM, G_DIM);        // wihT [4096 x 1024]
    gemm(0, z1_16, wBig1, b4096, pre, M_DIM, G_DIM, D_DIM);
    packT(wBig2, F32(WHH), G_DIM, D_DIM, G_DIM);        // whhT [4096 x 1024]
    lstm_scan_kernel<<<1, 1024, 0, stream>>>(pre, wBig2, hsbuf);

    // sig = mlp_sig(fc_sig, sigmoid(hs))
    packM(1, sh16, hsbuf, M_DIM, D_DIM, D_DIM);
    mlp3(sh16, D_DIM, D_DIM,
         F32(SG + 0), F32(SG + 1), F32(SG + 2), F32(SG + 3), F32(SG + 4), F32(SG + 5),
         0, t1);

    // mu = mlp_sig(fc_mu, z1)
    mlp3(z1_16, D_DIM, D_DIM,
         F32(MU + 0), F32(MU + 1), F32(MU + 2), F32(MU + 3), F32(MU + 4), F32(MU + 5),
         0, t2);

    // nuT = mlp_sig(nu, z1.T);  nu = nuT.T
    mlp3(z1T16, M_DIM, M_DIM,
         F32(NU + 0), F32(NU + 1), F32(NU + 2), F32(NU + 3), F32(NU + 4), F32(NU + 5),
         0, t3);
    packT(nu16, t3, M_DIM, D_DIM, M_DIM);

    // f = mlp_sig(f, nu)   (2 layers)
    packWT(wA, F32(F_W0), D_DIM, HID);
    gemm(1, nu16, wA, F32(F_B0), hf32, M_DIM, HID, D_DIM);
    packM(0, h16, hf32, M_DIM, HID, HID);
    packWT(wC, F32(F_W1), HID, D_DIM);
    gemm(0, h16, wC, F32(F_B1), t0, M_DIM, D_DIM, HID);

    // z2 = mu*f + x2*exp(sig)  (in place);  log_jacob += rowsum(sig)
    flow_out_kernel<<<(NHALF + 255) / 256, 256, 0, stream>>>(xbuf, t2, t0, t1, c2);
    rowsum_add_kernel<<<M_DIM, 256, 0, stream>>>(out_logjac, t1, D_DIM);
  }

  // --- 8. log_pz from final z ---
  logpz_kernel<<<M_DIM, 256, 0, stream>>>(out_logpz, xbuf);

  // --- 9. final x -> d_out ---
  hipMemcpyAsync(out_x, xbuf, (size_t)M_DIM * H_DIM * sizeof(float),
                 hipMemcpyDeviceToDevice, stream);
}
